// GINe_4569845203336
// MI455X (gfx1250) — matrix operand order repaired and checked by reading, not compiled, not run on hardware
//
#include <hip/hip_runtime.h>
#include <math.h>

typedef _Float16 half_t;
typedef __attribute__((ext_vector_type(16))) _Float16 v16h;
typedef __attribute__((ext_vector_type(8)))  _Float16 v8h;
typedef __attribute__((ext_vector_type(8)))  float    v8f;

#define HH 66

// GEMM epilogue modes
#define GM_RELU     1
#define GM_ACC_HALF 2   // Cf[idx] += 0.5f * (acc + bias)

// ---------------------------------------------------------------------------
// WMMA GEMM v2: C[M x (TN*16)] = A[M x Kp](f16, K zero-padded) @ Bt^T + bias
//   Bt[n*Kp + k] (transposed weight, Np = TN*16 rows, zero-padded) is staged
//   into LDS once per block; each wave owns one 16-row M tile and computes
//   ALL TN column tiles, so A is read from HBM exactly once.
//   Cf (optional): fp32 out, cols n < Nf, leading dim ldc.
//   Ch (optional): f16 out, all TN*16 cols stored, leading dim ldch.
// ---------------------------------------------------------------------------
template <int TN>
__global__ void wmma_gemm_kernel(const half_t* __restrict__ A,
                                 const half_t* __restrict__ Bt,
                                 const float*  __restrict__ bias,
                                 float* __restrict__ Cf, int ldc, int Nf,
                                 half_t* __restrict__ Ch, int ldch,
                                 int M, int Kp, int mode)
{
    extern __shared__ char smem[];
    half_t* Bsh = (half_t*)smem;

    // ---- cooperative LDS fill of the whole B panel (TN*16 x Kp halves) ----
    {
        const int nDwords = (TN * 16 * Kp) >> 1;          // 2 halves per dword
        const uint* s = (const uint*)Bt;
        uint* d = (uint*)Bsh;
        for (int i = threadIdx.x; i < nDwords; i += 256) d[i] = s[i];
    }
    __syncthreads();

    const int wave = threadIdx.x >> 5;
    const int lane = threadIdx.x & 31;
    const int tM   = blockIdx.x * 8 + wave;               // M tile for this wave
    if (tM >= (M >> 4)) return;                           // wave-uniform exit
    const int nl = lane & 15;                             // A row / B,C col in tile
    const int hf = lane >> 4;                             // lane half: K runs / C rows

    const half_t* Arow = A + (size_t)((tM << 4) + nl) * Kp;

    v8f acc[TN];
#pragma unroll
    for (int t = 0; t < TN; ++t) acc[t] = (v8f){};

    for (int k = 0; k < Kp; k += 32) {
        // A fragment: lane half selects 16-K run K[hf*16 .. +16)
        v16h a = *(const v16h*)(Arow + k + hf * 16);
#pragma unroll
        for (int t = 0; t < TN; ++t) {
            // B fragment from LDS: lane holds column n, half selects 16-K run
            v16h b = *(const v16h*)(Bsh + (size_t)((t << 4) + nl) * Kp + k + hf * 16);
            acc[t] = __builtin_amdgcn_wmma_f32_16x16x32_f16(false, a, false, b,
                                                            (short)0, acc[t],
                                                            false, false);
        }
    }

#pragma unroll
    for (int t = 0; t < TN; ++t) {
        const int n = (t << 4) + nl;
        const float bv = (bias && n < Nf) ? bias[n] : 0.0f;
#pragma unroll
        for (int r = 0; r < 8; ++r) {
            const int mr = (tM << 4) + (hf << 3) + r;     // C row = r + 8*half
            float v = acc[t][r] + bv;
            if (mode & GM_RELU) v = fmaxf(v, 0.0f);
            if (Cf && n < Nf) {
                size_t idx = (size_t)mr * ldc + n;
                if (mode & GM_ACC_HALF) Cf[idx] = Cf[idx] + 0.5f * v;
                else                    Cf[idx] = v;
            }
            if (Ch) Ch[(size_t)mr * ldch + n] = (half_t)v;
        }
    }
}

// A-side fp32 -> f16 with K zero-padding. blockDim=(Kp, rowsPerBlock)
__global__ void convert_pad_a(const float* __restrict__ A, half_t* __restrict__ Ah,
                              int K, int Kp)
{
    const int j = threadIdx.x;
    const long m = (long)blockIdx.x * blockDim.y + threadIdx.y;
    const float v = (j < K) ? A[m * K + j] : 0.0f;
    Ah[m * Kp + j] = (half_t)v;
}

// Weight [K x N] row-major fp32 -> transposed, zero-padded f16 Bt[Np x Kp].
__global__ void convert_wT(const float* __restrict__ W, half_t* __restrict__ Bt,
                           int K, int N, int Kp, int Np)
{
    const int idx = blockIdx.x * blockDim.x + threadIdx.x;
    if (idx >= Kp * Np) return;
    const int n = idx / Kp, k = idx - n * Kp;
    const float v = (n < N && k < K) ? W[(size_t)k * N + n] : 0.0f;
    Bt[(size_t)n * Kp + k] = (half_t)v;
}

__global__ void zero_f32(float* __restrict__ p, long n)
{
    long i = (long)blockIdx.x * blockDim.x + threadIdx.x;
    const long stride = (long)gridDim.x * blockDim.x;
    for (; i < n; i += stride) p[i] = 0.0f;
}

// msg = relu(h[src] + ea); aggr[dst] += msg.  blockDim=(66,4).
__global__ void edge_aggregate(const float* __restrict__ h,
                               const half_t* __restrict__ ea,   // E x 80 (f16)
                               const int* __restrict__ src, const int* __restrict__ dst,
                               float* __restrict__ aggr, int E)
{
    const int c = threadIdx.x;
    const long e = (long)blockIdx.x * blockDim.y + threadIdx.y;
    if (e >= E) return;
    const int s = src[e], d = dst[e];
    float v = h[(long)s * HH + c] + (float)ea[e * 80 + c];
    v = fmaxf(v, 0.0f);
    atomicAdd(&aggr[(long)d * HH + c], v);
}

// hz = (f16)(h + aggr), padded to 96 cols.  blockDim=(96,2).
__global__ void hz_pack(const float* __restrict__ h, const float* __restrict__ aggr,
                        half_t* __restrict__ out)
{
    const int j = threadIdx.x;
    const long m = (long)blockIdx.x * blockDim.y + threadIdx.y;
    float v = 0.0f;
    if (j < HH) { const long idx = m * HH + j; v = h[idx] + aggr[idx]; }
    out[m * 96 + j] = (half_t)v;
}

// BatchNorm stats via per-column atomic reduction. blockDim=(66,4).
__global__ void bn_stats(const float* __restrict__ z, float* __restrict__ sums,
                         float* __restrict__ sumsq, int N)
{
    const int c = threadIdx.x;
    float s = 0.0f, s2 = 0.0f;
    for (long r = (long)blockIdx.x * blockDim.y + threadIdx.y; r < N;
         r += (long)gridDim.x * blockDim.y) {
        const float v = z[r * HH + c];
        s += v; s2 += v * v;
    }
    atomicAdd(&sums[c], s);
    atomicAdd(&sumsq[c], s2);
}

// h = (h + relu(BN(z))) / 2. blockDim=(66,4).
__global__ void bn_update_h(const float* __restrict__ z, float* __restrict__ h,
                            const float* __restrict__ sums, const float* __restrict__ sumsq,
                            const float* __restrict__ gamma, const float* __restrict__ beta,
                            int N)
{
    const int c = threadIdx.x;
    const long r = (long)blockIdx.x * blockDim.y + threadIdx.y;
    const float mean = sums[c] / (float)N;
    const float var  = sumsq[c] / (float)N - mean * mean;
    const float inv  = rsqrtf(var + 1e-5f);
    const long idx = r * HH + c;
    const float zn = (z[idx] - mean) * inv * gamma[c] + beta[c];
    h[idx] = 0.5f * (h[idx] + fmaxf(zn, 0.0f));
}

// cat = [h[tsrc], h[tdst], tea] -> f16, K padded to 224. blockDim=224, grid=ET.
__global__ void build_cat(const float* __restrict__ h, const float* __restrict__ tea,
                          const int* __restrict__ tsrc, const int* __restrict__ tdst,
                          half_t* __restrict__ cat, int reluH)
{
    const long e = blockIdx.x;
    const int j = threadIdx.x;
    float v = 0.0f;
    if (j < 66)       v = h[(long)tsrc[e] * HH + j];
    else if (j < 132) v = h[(long)tdst[e] * HH + (j - 66)];
    else if (j < 198) v = tea[e * HH + (j - 132)];
    if (reluH && j < 132) v = fmaxf(v, 0.0f);
    cat[e * 224 + j] = (half_t)v;
}

// ---------------------------------------------------------------------------
extern "C" void kernel_launch(void* const* d_in, const int* in_sizes, int n_in,
                              void* d_out, int out_size, void* d_ws, size_t ws_size,
                              hipStream_t stream)
{
    (void)in_sizes; (void)n_in; (void)out_size; (void)ws_size;
    const int N = 100000, E = 1000000, ET = 200000, F = 64, D = 32, L = 2;

    const float* x       = (const float*)d_in[0];
    const int*   ei      = (const int*)  d_in[1];
    const float* eattr   = (const float*)d_in[2];
    const int*   eli     = (const int*)  d_in[3];
    const float* teattr  = (const float*)d_in[4];
    const float* node_w  = (const float*)d_in[5];
    const float* node_b  = (const float*)d_in[6];
    const float* edge_w  = (const float*)d_in[7];
    const float* edge_b  = (const float*)d_in[8];
    const float* conv_w1 = (const float*)d_in[9];
    const float* conv_b1 = (const float*)d_in[10];
    const float* conv_w2 = (const float*)d_in[11];
    const float* conv_b2 = (const float*)d_in[12];
    const float* bn_g    = (const float*)d_in[13];
    const float* bn_b    = (const float*)d_in[14];
    const float* emlp_w1 = (const float*)d_in[15];
    const float* emlp_b1 = (const float*)d_in[16];
    const float* emlp_w2 = (const float*)d_in[17];
    const float* emlp_b2 = (const float*)d_in[18];
    const float* mlp_w1  = (const float*)d_in[19];
    const float* mlp_b1  = (const float*)d_in[20];
    const float* mlp_w2  = (const float*)d_in[21];
    const float* mlp_b2  = (const float*)d_in[22];
    const float* mlp_w3  = (const float*)d_in[23];
    const float* mlp_b3  = (const float*)d_in[24];
    const int* src = ei,  * dst = ei + E;
    const int* tsrc = eli, * tdst = eli + ET;

    // bump allocator over workspace
    char* base = (char*)d_ws;
    size_t off = 0;
    auto alloc = [&](size_t bytes) -> char* {
        char* p = base + off;
        off = (off + bytes + 255) & ~(size_t)255;
        return p;
    };

    // persistent buffers
    float*  h     = (float*) alloc((size_t)N  * HH * 4);
    half_t* ea16  = (half_t*)alloc((size_t)E  * 80 * 2);   // edge embeddings, f16
    float*  tea   = (float*) alloc((size_t)ET * HH * 4);
    float*  aggr  = (float*) alloc((size_t)N  * HH * 4);
    float*  z     = (float*) alloc((size_t)N  * HH * 4);
    float*  sums  = (float*) alloc(132 * 4);               // [sums(66) | sumsq(66)]
    float*  sumsq = sums + 66;
    half_t* Wt0   = (half_t*)alloc((size_t)96 * 224 * 2);
    half_t* Wt1   = (half_t*)alloc((size_t)96 * 224 * 2);
    const size_t arena = off;

    auto gemm = [&](const half_t* A, const half_t* Bt, const float* bias,
                    float* Cf, int ldc, int Nf, half_t* Ch, int ldch,
                    int M, int Kp, int tilesN, int mode) {
        const int blocks = ((M >> 4) + 7) / 8;
        const size_t shmem = (size_t)tilesN * 16 * Kp * sizeof(half_t);
        switch (tilesN) {
        case 1: wmma_gemm_kernel<1><<<blocks, 256, shmem, stream>>>(
                    A, Bt, bias, Cf, ldc, Nf, Ch, ldch, M, Kp, mode); break;
        case 2: wmma_gemm_kernel<2><<<blocks, 256, shmem, stream>>>(
                    A, Bt, bias, Cf, ldc, Nf, Ch, ldch, M, Kp, mode); break;
        case 4: wmma_gemm_kernel<4><<<blocks, 256, shmem, stream>>>(
                    A, Bt, bias, Cf, ldc, Nf, Ch, ldch, M, Kp, mode); break;
        case 5: wmma_gemm_kernel<5><<<blocks, 256, shmem, stream>>>(
                    A, Bt, bias, Cf, ldc, Nf, Ch, ldch, M, Kp, mode); break;
        default: wmma_gemm_kernel<6><<<blocks, 256, shmem, stream>>>(
                    A, Bt, bias, Cf, ldc, Nf, Ch, ldch, M, Kp, mode); break;
        }
    };
    auto wconv = [&](const float* W, half_t* Bt, int K, int Ncols, int Kp, int Np) {
        const int tot = Kp * Np;
        convert_wT<<<(tot + 255) / 256, 256, 0, stream>>>(W, Bt, K, Ncols, Kp, Np);
    };

    // ---------------- Phase 0: embeddings ----------------
    off = arena;
    half_t* xh   = (half_t*)alloc((size_t)N  * 64 * 2);
    half_t* eah  = (half_t*)alloc((size_t)E  * 32 * 2);
    half_t* teah = (half_t*)alloc((size_t)ET * 32 * 2);

    convert_pad_a<<<N  / 4, dim3(64, 4), 0, stream>>>(x,      xh,   F, 64);
    convert_pad_a<<<E  / 4, dim3(32, 4), 0, stream>>>(eattr,  eah,  D, 32);
    convert_pad_a<<<ET / 4, dim3(32, 4), 0, stream>>>(teattr, teah, D, 32);
    wconv(node_w, Wt0, 64, 66, 64, 80);
    wconv(edge_w, Wt1, 32, 66, 32, 80);

    gemm(xh,   Wt0, node_b, h,   HH, 66, nullptr, 0, N,  64, 5, 0);  // h = x@Wn+b
    gemm(eah,  Wt1, edge_b, nullptr, 0, 66, ea16, 80, E,  32, 5, 0); // ea (f16, E x 80)
    gemm(teah, Wt1, edge_b, tea, HH, 66, nullptr, 0, ET, 32, 5, 0);  // tea

    // ---------------- GINE layers ----------------
    for (int i = 0; i < L; ++i) {
        off = arena;
        half_t* hzh  = (half_t*)alloc((size_t)N  * 96  * 2);
        half_t* t1h  = (half_t*)alloc((size_t)N  * 96  * 2);
        half_t* cath = (half_t*)alloc((size_t)ET * 224 * 2);
        half_t* u1h  = (half_t*)alloc((size_t)ET * 96  * 2);

        // message + scatter-add
        zero_f32<<<2048, 256, 0, stream>>>(aggr, (long)N * HH);
        edge_aggregate<<<E / 4, dim3(66, 4), 0, stream>>>(h, ea16, src, dst, aggr, E);
        hz_pack<<<N / 2, dim3(96, 2), 0, stream>>>(h, aggr, hzh);

        // conv MLP: relu((h+aggr)@W1+b1)@W2+b2
        wconv(conv_w1 + (size_t)i * 66 * 66, Wt0, 66, 66, 96, 96);
        wconv(conv_w2 + (size_t)i * 66 * 66, Wt1, 66, 66, 96, 80);
        gemm(hzh, Wt0, conv_b1 + i * 66, nullptr, 0, 66, t1h, 96, N, 96, 6, GM_RELU);
        gemm(t1h, Wt1, conv_b2 + i * 66, z, HH, 66, nullptr, 0, N, 96, 5, 0);

        // BatchNorm + residual update of h
        zero_f32<<<1, 256, 0, stream>>>(sums, 132);
        bn_stats<<<512, dim3(66, 4), 0, stream>>>(z, sums, sumsq, N);
        bn_update_h<<<N / 4, dim3(66, 4), 0, stream>>>(z, h, sums, sumsq,
                                                       bn_g + i * 66, bn_b + i * 66, N);

        // edge update: tea += 0.5 * (relu(cat@W1+b1)@W2+b2)
        build_cat<<<ET, 224, 0, stream>>>(h, tea, tsrc, tdst, cath, 0);
        wconv(emlp_w1 + (size_t)i * 198 * 66, Wt0, 198, 66, 224, 96);
        wconv(emlp_w2 + (size_t)i * 66 * 66,  Wt1, 66,  66, 96,  80);
        gemm(cath, Wt0, emlp_b1 + i * 66, nullptr, 0, 66, u1h, 96, ET, 224, 6, GM_RELU);
        gemm(u1h,  Wt1, emlp_b2 + i * 66, tea, HH, 66, nullptr, 0, ET, 96, 5, GM_ACC_HALF);
    }

    // ---------------- Classifier ----------------
    off = arena;
    half_t* feat = (half_t*)alloc((size_t)ET * 224 * 2);
    half_t* o1h  = (half_t*)alloc((size_t)ET * 64  * 2);
    half_t* o2h  = (half_t*)alloc((size_t)ET * 32  * 2);

    build_cat<<<ET, 224, 0, stream>>>(h, tea, tsrc, tdst, feat, 1);  // relu on h parts
    wconv(mlp_w1, Wt0, 198, 50, 224, 64);
    gemm(feat, Wt0, mlp_b1, nullptr, 0, 50, o1h, 64, ET, 224, 4, GM_RELU);
    wconv(mlp_w2, Wt1, 50, 25, 64, 32);
    gemm(o1h, Wt1, mlp_b2, nullptr, 0, 25, o2h, 32, ET, 64, 2, GM_RELU);
    wconv(mlp_w3, Wt0, 25, 2, 32, 16);
    gemm(o2h, Wt0, mlp_b3, (float*)d_out, 2, 2, nullptr, 0, ET, 32, 1, 0);
}
